// NDNRelation_14499809591375
// MI455X (gfx1250) — matrix operand order repaired and checked by reading, not compiled, not run on hardware
//
#include <hip/hip_runtime.h>
#include <hip/hip_bf16.h>

typedef unsigned short u16;
typedef __attribute__((ext_vector_type(16))) __bf16 bf16x16;
typedef __attribute__((ext_vector_type(8)))  float  f32x8;

#define N_OBJ   30000
#define T_EDGES 100000
#define ALPHA   0.2f
#define BN_EPS  1e-3f

__device__ __forceinline__ u16 f2bf(float f) {
  unsigned int u = __float_as_uint(f);
  unsigned int r = u + 0x7FFFu + ((u >> 16) & 1u);
  return (u16)(r >> 16);
}
__device__ __forceinline__ float lrelu(float x) { return x > 0.f ? x : ALPHA * x; }

// ---------------------------------------------------------------------------
// Weight pack: W[K,N] f32 -> Wp bf16 in WMMA B-fragment order, N padded to
// Npad (multiple of 128) with zeros so the GEMM K-loop needs no N guards.
// Tile (nt,kt) = 512 elements (1KB). Within tile: lane l (0..31) owns 16
// contiguous bf16:  Wp[...|l|e] = W[kt*32 + 16*(l>>4) + e][nt*16 + (l&15)]
// => GEMM reads a whole lane fragment as two 16B global loads.
// ---------------------------------------------------------------------------
__global__ __launch_bounds__(256) void k_pack_w(const float* __restrict__ W,
                                                u16* __restrict__ Wp,
                                                int K, int N, int Npad) {
  long idx = (long)blockIdx.x * 256 + threadIdx.x;
  long total = (long)K * Npad;
  if (idx >= total) return;
  int e  = (int)(idx & 15);
  int l  = (int)(idx >> 4) & 31;
  long rest = idx >> 9;
  int ktc = K >> 5;
  int kt = (int)(rest % ktc);
  int nt = (int)(rest / ktc);
  int k = kt * 32 + 16 * (l >> 4) + e;
  int n = nt * 16 + (l & 15);
  u16 v = 0;
  if (n < N) v = f2bf(W[(long)k * N + n]);
  Wp[idx] = v;
}

// ---------------------------------------------------------------------------
// WMMA bf16 GEMM:  C[M,N] = A[M,K] * Bp(packed, Npad cols) + bias[N]
// block = 256 thr = 8 waves; tile 64(M) x 128(N) x 32(K)
// wave (wm 0..3, wn 0..1): rows 16*wm..+15, cols 64*wn..+63 (4 wmma acc)
// A staged via LDS (shared by the wn pair); B fragments read directly from
// the packed weight buffer (L2-resident) as 2x b128 per fragment.
// Inner loop is branch-free: 8 global b128 loads, then 4 wmma.
// ---------------------------------------------------------------------------
__global__ __launch_bounds__(256) void k_gemm_bias(
    const u16* __restrict__ A, const u16* __restrict__ Bp,
    const float* __restrict__ bias, float* __restrict__ C,
    int M, int N, int K)
{
  constexpr int LDA = 36;   // 64x32 A tile, padded stride (elements)
  __shared__ u16 sA[64 * LDA];

  const int tid  = threadIdx.x;
  const int lane = tid & 31;
  const int l15  = lane & 15;
  const int kh   = lane >> 4;
  const int wave = tid >> 5;
  const int wm   = wave >> 1;
  const int wn   = wave & 1;
  const long m0 = (long)blockIdx.y * 64;
  const long n0 = (long)blockIdx.x * 128;
  const int ktc = K >> 5;

  f32x8 acc[4] = {};

  const int arow = tid >> 2;          // 0..63
  const int acol = (tid & 3) * 8;     // 0,8,16,24

  const long ntbase = (n0 >> 4) + 4 * wn;
  // per-fragment base pointers (advance by 512 elements per K-step)
  const u16* pb0 = Bp + (((ntbase + 0) * ktc) << 9) + lane * 16;
  const u16* pb1 = Bp + (((ntbase + 1) * ktc) << 9) + lane * 16;
  const u16* pb2 = Bp + (((ntbase + 2) * ktc) << 9) + lane * 16;
  const u16* pb3 = Bp + (((ntbase + 3) * ktc) << 9) + lane * 16;

  for (int k0 = 0; k0 < K; k0 += 32) {
    __syncthreads();
    { // A tile: 64x32, one 16B chunk per thread
      uint4 v = make_uint4(0u, 0u, 0u, 0u);
      long gr = m0 + arow;
      if (gr < M) {
        const u16* gp = A + gr * (long)K + k0 + acol;
        v = *(const uint4*)gp;
        if (k0 + 32 < K) __builtin_prefetch(gp + 32, 0, 3);
      }
      unsigned int* d = (unsigned int*)(sA + arow * LDA + acol);
      d[0] = v.x; d[1] = v.y; d[2] = v.z; d[3] = v.w;
    }
    __syncthreads();

    // A fragment (16x32 bf16): VGPR r lanes0-15 K=2r,2r+1 ; lanes16-31 K=+16 ; VGPR4-7 K=+8
    union { unsigned int u[8]; bf16x16 v; } fa;
    const u16* pa = sA + (16 * wm + l15) * LDA;
#pragma unroll
    for (int r = 0; r < 4; ++r) {
      fa.u[r]     = *(const unsigned int*)(pa + 2 * r + 16 * kh);
      fa.u[4 + r] = *(const unsigned int*)(pa + 8 + 2 * r + 16 * kh);
    }

    // All B fragments first (one clause of 8 b128), then 4 independent wmma.
    union { uint4 q[2]; bf16x16 v; } fb0, fb1, fb2, fb3;
    fb0.q[0] = *(const uint4*)(pb0);  fb0.q[1] = *(const uint4*)(pb0 + 8);
    fb1.q[0] = *(const uint4*)(pb1);  fb1.q[1] = *(const uint4*)(pb1 + 8);
    fb2.q[0] = *(const uint4*)(pb2);  fb2.q[1] = *(const uint4*)(pb2 + 8);
    fb3.q[0] = *(const uint4*)(pb3);  fb3.q[1] = *(const uint4*)(pb3 + 8);
    pb0 += 512; pb1 += 512; pb2 += 512; pb3 += 512;

    acc[0] = __builtin_amdgcn_wmma_f32_16x16x32_bf16(false, fa.v, false, fb0.v, (short)0, acc[0], false, false);
    acc[1] = __builtin_amdgcn_wmma_f32_16x16x32_bf16(false, fa.v, false, fb1.v, (short)0, acc[1], false, false);
    acc[2] = __builtin_amdgcn_wmma_f32_16x16x32_bf16(false, fa.v, false, fb2.v, (short)0, acc[2], false, false);
    acc[3] = __builtin_amdgcn_wmma_f32_16x16x32_bf16(false, fa.v, false, fb3.v, (short)0, acc[3], false, false);
  }

  // C layout: VGPR v, lanes0-15 -> M=v, lanes16-31 -> M=8+v ; N = lane&15
#pragma unroll
  for (int j = 0; j < 4; ++j) {
    long col = n0 + 64 * wn + 16 * j + l15;
    if (col < N) {
      float bv = bias ? bias[col] : 0.f;
#pragma unroll
      for (int v = 0; v < 8; ++v) {
        long row = m0 + 16 * wm + 8 * kh + v;
        if (row < M) C[row * (long)N + col] = acc[j][v] + bv;
      }
    }
  }
}

// ---------------------------------------------------------------------------
// helpers
// ---------------------------------------------------------------------------
__global__ __launch_bounds__(256) void k_f32_to_bf16(const float* __restrict__ in,
                                                     u16* __restrict__ out, long n) {
  long i = (long)blockIdx.x * 256 + threadIdx.x;
  if (i < n) out[i] = f2bf(in[i]);
}

__global__ __launch_bounds__(256) void k_fill0(float* __restrict__ p, long n) {
  long i = (long)blockIdx.x * 256 + threadIdx.x;
  if (i < n) p[i] = 0.f;
}

// column mean / rsqrt(var+eps): 64 cols per block, 4 row-lanes
__global__ __launch_bounds__(256) void k_colstats(const float* __restrict__ X, int M, int N,
                                                  float* __restrict__ mean, float* __restrict__ rstd) {
  __shared__ float ssum[256], ssq[256];
  int cl = threadIdx.x & 63, rl = threadIdx.x >> 6;
  int c = blockIdx.x * 64 + cl;
  float s = 0.f, q = 0.f;
  if (c < N)
    for (long r = rl; r < M; r += 4) { float v = X[r * (long)N + c]; s += v; q += v * v; }
  ssum[threadIdx.x] = s; ssq[threadIdx.x] = q;
  __syncthreads();
  if (rl == 0 && c < N) {
    for (int k = 1; k < 4; ++k) { s += ssum[k * 64 + cl]; q += ssq[k * 64 + cl]; }
    float m = s / (float)M;
    float v = q / (float)M - m * m;
    mean[c] = m;
    rstd[c] = rsqrtf(fmaxf(v, 0.f) + BN_EPS);
  }
}

__global__ __launch_bounds__(256) void k_bn_bf16(const float* __restrict__ X,
    const float* __restrict__ mean, const float* __restrict__ rstd,
    const float* __restrict__ g, const float* __restrict__ be,
    u16* __restrict__ out, long n, int N) {
  long i = (long)blockIdx.x * 256 + threadIdx.x;
  if (i >= n) return;
  int c = (int)(i % N);
  out[i] = f2bf(lrelu(g[c] * (X[i] - mean[c]) * rstd[c] + be[c]));
}

__global__ __launch_bounds__(256) void k_bn_f32(const float* __restrict__ X,
    const float* __restrict__ mean, const float* __restrict__ rstd,
    const float* __restrict__ g, const float* __restrict__ be,
    float* __restrict__ out, long n, int N) {
  long i = (long)blockIdx.x * 256 + threadIdx.x;
  if (i >= n) return;
  int c = (int)(i % N);
  out[i] = lrelu(g[c] * (X[i] - mean[c]) * rstd[c] + be[c]);
}

// build t = [obj[s] | pred | obj[o]] as bf16
__global__ __launch_bounds__(128) void k_gather_concat(
    const float* __restrict__ obj, const float* __restrict__ pred,
    const int* __restrict__ sidx, const int* __restrict__ oidx,
    u16* __restrict__ out, int din) {
  int i = blockIdx.x;
  int s = sidx[i], o = oidx[i];
  long base = (long)i * (3 * din);
  for (int c = threadIdx.x; c < din; c += 128) {
    out[base + c]            = f2bf(obj[(long)s * din + c]);
    out[base + din + c]      = f2bf(pred[(long)i * din + c]);
    out[base + 2 * din + c]  = f2bf(obj[(long)o * din + c]);
  }
}

// pooled.at[s]+=new_s ; pooled.at[o]+=new_o ; counts ; copy new_pred
__global__ __launch_bounds__(128) void k_scatter(
    const float* __restrict__ nt, const int* __restrict__ sidx, const int* __restrict__ oidx,
    float* __restrict__ pooled, float* __restrict__ counts, float* __restrict__ npred,
    int H, int Dout, int Dn) {
  int i = blockIdx.x;
  int s = sidx[i], o = oidx[i];
  long b = (long)i * Dn;
  for (int c = threadIdx.x; c < H; c += 128) {
    atomicAdd(&pooled[(long)s * H + c], nt[b + c]);
    atomicAdd(&pooled[(long)o * H + c], nt[b + H + Dout + c]);
  }
  for (int c = threadIdx.x; c < Dout; c += 128)
    npred[(long)i * Dout + c] = nt[b + H + c];
  if (threadIdx.x == 0) { atomicAdd(&counts[s], 1.f); atomicAdd(&counts[o], 1.f); }
}

__global__ __launch_bounds__(256) void k_pooled_div(float* __restrict__ p,
                                                    const float* __restrict__ counts,
                                                    long n, int H) {
  long i = (long)blockIdx.x * 256 + threadIdx.x;
  if (i >= n) return;
  p[i] /= fmaxf(counts[i / H], 1.f);
}

// z_mu / z_var / z = eps*exp(z_var/2)+z_mu   (K=32, N=32)
__global__ __launch_bounds__(32) void k_zmuvar(const float* __restrict__ og,
    const float* __restrict__ Wm, const float* __restrict__ bm,
    const float* __restrict__ Wv, const float* __restrict__ bv,
    const float* __restrict__ eps,
    float* __restrict__ zmu, float* __restrict__ zvar, float* __restrict__ z) {
  int i = blockIdx.x, c = threadIdx.x;
  float mu = bm[c], va = bv[c];
  for (int k = 0; k < 32; ++k) {
    float x = og[(long)i * 32 + k];
    mu += x * Wm[k * 32 + c];
    va += x * Wv[k * 32 + c];
  }
  long idx = (long)i * 32 + c;
  zmu[idx] = mu;
  zvar[idx] = va;
  z[idx] = eps[idx] * __expf(va * 0.5f) + mu;
}

// new_obj = leaky([obj | z] @ W(96x64) + b)
__global__ __launch_bounds__(64) void k_nodeemb(const float* __restrict__ obj,
    const float* __restrict__ z, const float* __restrict__ W,
    const float* __restrict__ b, float* __restrict__ out) {
  int i = blockIdx.x, c = threadIdx.x;
  float a = b[c];
  for (int k = 0; k < 64; ++k) a += obj[(long)i * 64 + k] * W[k * 64 + c];
  for (int k = 0; k < 32; ++k) a += z[(long)i * 32 + k] * W[(64 + k) * 64 + c];
  out[(long)i * 64 + c] = lrelu(a);
}

__global__ __launch_bounds__(128) void k_softmax_argmax(const float* __restrict__ lg,
                                                        float* __restrict__ cls,
                                                        float* __restrict__ np) {
  long i = (long)blockIdx.x * 128 + threadIdx.x;
  if (i >= T_EDGES) return;
  float v[16], mx = -1e30f;
  for (int j = 0; j < 16; ++j) { v[j] = lg[i * 16 + j]; mx = fmaxf(mx, v[j]); }
  float sm = 0.f;
  for (int j = 0; j < 16; ++j) { v[j] = __expf(v[j] - mx); sm += v[j]; }
  float inv = 1.f / sm;
  int am = 0; float bvv = -1.f;
  for (int j = 0; j < 16; ++j) {
    float p = v[j] * inv;
    cls[i * 16 + j] = p;
    if (p > bvv) { bvv = p; am = j; }
  }
  np[i] = (float)am;
}

// ---------------------------------------------------------------------------
// host orchestration
// ---------------------------------------------------------------------------
struct Bufs {
  u16 *bfA, *bfH, *wbuf;
  float *xbuf, *predbuf, *objbuf, *pooled, *counts, *meanb, *rstdb, *zbuf;
};

static inline int cdl(long a, long b) { return (int)((a + b - 1) / b); }

// one 2-layer dense+BN+leaky MLP; A0 bf16 [M,d0]; params at d_in[widx..widx+7]
static void run_mlp(const u16* A0, int M, int d0, int d1, int d2,
                    void* const* d_in, int widx, Bufs& B,
                    u16* out_bf, float* out_f32, hipStream_t st) {
  const float* W0 = (const float*)d_in[widx + 0];
  const float* b0 = (const float*)d_in[widx + 1];
  const float* g0 = (const float*)d_in[widx + 2];
  const float* e0 = (const float*)d_in[widx + 3];
  const float* W1 = (const float*)d_in[widx + 4];
  const float* b1 = (const float*)d_in[widx + 5];
  const float* g1 = (const float*)d_in[widx + 6];
  const float* e1 = (const float*)d_in[widx + 7];

  int np1 = cdl(d1, 128) * 128;
  int np2 = cdl(d2, 128) * 128;

  long n = (long)d0 * np1;
  k_pack_w<<<cdl(n, 256), 256, 0, st>>>(W0, B.wbuf, d0, d1, np1);
  dim3 gr1(np1 / 128, cdl(M, 64));
  k_gemm_bias<<<gr1, 256, 0, st>>>(A0, B.wbuf, b0, B.xbuf, M, d1, d0);
  k_colstats<<<cdl(d1, 64), 256, 0, st>>>(B.xbuf, M, d1, B.meanb, B.rstdb);
  n = (long)M * d1;
  k_bn_bf16<<<cdl(n, 256), 256, 0, st>>>(B.xbuf, B.meanb, B.rstdb, g0, e0, B.bfH, n, d1);

  n = (long)d1 * np2;
  k_pack_w<<<cdl(n, 256), 256, 0, st>>>(W1, B.wbuf, d1, d2, np2);
  dim3 gr2(np2 / 128, cdl(M, 64));
  k_gemm_bias<<<gr2, 256, 0, st>>>(B.bfH, B.wbuf, b1, B.xbuf, M, d2, d1);
  k_colstats<<<cdl(d2, 64), 256, 0, st>>>(B.xbuf, M, d2, B.meanb, B.rstdb);
  n = (long)M * d2;
  if (out_bf)
    k_bn_bf16<<<cdl(n, 256), 256, 0, st>>>(B.xbuf, B.meanb, B.rstdb, g1, e1, out_bf, n, d2);
  else
    k_bn_f32<<<cdl(n, 256), 256, 0, st>>>(B.xbuf, B.meanb, B.rstdb, g1, e1, out_f32, n, d2);
}

static void gconv(const float* obj, const float* pred, int din, int h, int dout,
                  int pidx, void* const* d_in, const int* sidx, const int* oidx,
                  Bufs& B, float* obj_out, float* pred_out, hipStream_t st) {
  int Dn = 2 * h + dout;
  k_gather_concat<<<T_EDGES, 128, 0, st>>>(obj, pred, sidx, oidx, B.bfA, din);
  // net1 -> new_t (f32, in-place BN in xbuf)
  run_mlp(B.bfA, T_EDGES, 3 * din, h, Dn, d_in, pidx, B, nullptr, B.xbuf, st);
  // pooling
  k_fill0<<<cdl((long)N_OBJ * h, 256), 256, 0, st>>>(B.pooled, (long)N_OBJ * h);
  k_fill0<<<cdl(N_OBJ, 256), 256, 0, st>>>(B.counts, N_OBJ);
  k_scatter<<<T_EDGES, 128, 0, st>>>(B.xbuf, sidx, oidx, B.pooled, B.counts, pred_out, h, dout, Dn);
  k_pooled_div<<<cdl((long)N_OBJ * h, 256), 256, 0, st>>>(B.pooled, B.counts, (long)N_OBJ * h, h);
  k_f32_to_bf16<<<cdl((long)N_OBJ * h, 256), 256, 0, st>>>(B.pooled, B.bfA, (long)N_OBJ * h);
  // net2 -> new_obj (f32)
  run_mlp(B.bfA, N_OBJ, h, h, dout, d_in, pidx + 8, B, nullptr, obj_out, st);
}

extern "C" void kernel_launch(void* const* d_in, const int* in_sizes, int n_in,
                              void* d_out, int out_size, void* d_ws, size_t ws_size,
                              hipStream_t stream) {
  (void)in_sizes; (void)out_size; (void)ws_size;
  const float* obj_vecs  = (const float*)d_in[0];
  const float* pred_gt   = (const float*)d_in[1];
  const float* pred_vecs = (const float*)d_in[2];
  const float* eps       = (const float*)d_in[3];
  const int*   sidx      = (const int*)d_in[n_in - 2];
  const int*   oidx      = (const int*)d_in[n_in - 1];
  // params flattened in sorted-key pytree order: gc, gp, h_pred, node_emb, z_mu, z_var
  const int PB = 4;

  char* w = (char*)d_ws;
  size_t off = 0;
  auto bump = [&](size_t bytes) -> void* {
    void* p = w + off;
    off = (off + bytes + 255) & ~(size_t)255;
    return p;
  };
  Bufs B;
  B.bfA     = (u16*)bump((size_t)T_EDGES * 512 * 2);
  B.bfH     = (u16*)bump((size_t)T_EDGES * 512 * 2);
  B.wbuf    = (u16*)bump((size_t)512 * 1152 * 2);
  B.xbuf    = (float*)bump((size_t)T_EDGES * 1152 * 4);
  B.predbuf = (float*)bump((size_t)T_EDGES * 128 * 4);
  B.objbuf  = (float*)bump((size_t)N_OBJ * 128 * 4);
  B.pooled  = (float*)bump((size_t)N_OBJ * 512 * 4);
  B.counts  = (float*)bump((size_t)N_OBJ * 4);
  B.meanb   = (float*)bump(1152 * 4);
  B.rstdb   = (float*)bump(1152 * 4);
  B.zbuf    = (float*)bump((size_t)N_OBJ * 32 * 4);

  float* out      = (float*)d_out;
  float* out_zmu  = out;
  float* out_zvar = out + (size_t)N_OBJ * 32;
  float* out_cls  = out_zvar + (size_t)N_OBJ * 32;
  float* out_np   = out_cls + (size_t)T_EDGES * 16;

  const int gc_dims[4][3] = {{64,512,128},{128,512,128},{128,512,128},{128,128,32}};
  const int gp_dims[4][3] = {{64,512,128},{128,512,128},{128,512,128},{128,128,128}};

  // --- gc stack on (obj_vecs, pred_gt_vecs) ---
  const float* obj = obj_vecs;
  const float* pred = pred_gt;
  for (int L = 0; L < 4; ++L) {
    gconv(obj, pred, gc_dims[L][0], gc_dims[L][1], gc_dims[L][2],
          PB + L * 16, d_in, sidx, oidx, B, B.objbuf, B.predbuf, stream);
    obj = B.objbuf; pred = B.predbuf;
  }

  // --- z_mu / z_var / reparameterize ---
  const float* Wm = (const float*)d_in[PB + 138];
  const float* bm = (const float*)d_in[PB + 139];
  const float* Wv = (const float*)d_in[PB + 140];
  const float* bv = (const float*)d_in[PB + 141];
  k_zmuvar<<<N_OBJ, 32, 0, stream>>>(B.objbuf, Wm, bm, Wv, bv, eps, out_zmu, out_zvar, B.zbuf);

  // --- node embedding: leaky([obj_vecs | z] @ Wn + bn) ---
  const float* Wn = (const float*)d_in[PB + 136];
  const float* bn = (const float*)d_in[PB + 137];
  k_nodeemb<<<N_OBJ, 64, 0, stream>>>(obj_vecs, B.zbuf, Wn, bn, B.objbuf);

  // --- gp stack on (new_obj, pred_vecs) ---
  obj = B.objbuf; pred = pred_vecs;
  for (int L = 0; L < 4; ++L) {
    gconv(obj, pred, gp_dims[L][0], gp_dims[L][1], gp_dims[L][2],
          PB + 64 + L * 16, d_in, sidx, oidx, B, B.objbuf, B.predbuf, stream);
    obj = B.objbuf; pred = B.predbuf;
  }

  // --- h_pred head [128,512,16] on new_pred (T x 128) ---
  k_f32_to_bf16<<<cdl((long)T_EDGES * 128, 256), 256, 0, stream>>>(B.predbuf, B.bfA, (long)T_EDGES * 128);
  run_mlp(B.bfA, T_EDGES, 128, 512, 16, d_in, PB + 128, B, nullptr, B.xbuf, stream);
  k_softmax_argmax<<<cdl(T_EDGES, 128), 128, 0, stream>>>(B.xbuf, out_cls, out_np);
}
